// DyGraphConv2d_43439299232098
// MI455X (gfx1250) — matrix operand order repaired and checked by reading, not compiled, not run on hardware
//
#include <hip/hip_runtime.h>
#include <math.h>

typedef __attribute__((ext_vector_type(16))) _Float16 v16h;
typedef __attribute__((ext_vector_type(8)))  _Float16 v8h;
typedef __attribute__((ext_vector_type(8)))  float    v8f;

#define B_   4
#define C_   192
#define N_   3136      // 56*56
#define KNN_ 9
#define CO_  384
#define CK_  384       // 2*C

// Forces all 6 fragments to be live at once: loads must finish before this
// point, WMMAs consume the asm's outputs -> loads cluster into one clause.
#define FRAG_BARRIER6(f) \
    asm volatile("" : "+v"(f[0]), "+v"(f[1]), "+v"(f[2]), \
                      "+v"(f[3]), "+v"(f[4]), "+v"(f[5]))

// -------------------------------------------------------------------------
// Prep 1: per-point transpose (B,C,N)->(B,N,C) fp32 and L2-normalized f16.
// -------------------------------------------------------------------------
__global__ void prep_points_kernel(const float* __restrict__ x,
                                   float* __restrict__ xt,
                                   _Float16* __restrict__ xnh) {
    int t = blockIdx.x * blockDim.x + threadIdx.x;
    if (t >= B_ * N_) return;
    int b = t / N_, n = t % N_;
    const float* xb = x + (size_t)b * C_ * N_ + n;   // stride N_ per channel
    float* row = xt + (size_t)t * C_;
    float ss = 0.f;
    for (int c = 0; c < C_; ++c) {
        float v = xb[(size_t)c * N_];
        ss += v * v;
        row[c] = v;
    }
    float s = 1.f / fmaxf(sqrtf(ss), 1e-12f);
    _Float16* hr = xnh + (size_t)t * C_;
    for (int c = 0; c < C_; ++c) hr[c] = (_Float16)(row[c] * s);
}

// -------------------------------------------------------------------------
// Prep 2: weights fp32 -> f16 (row-major 384x384; rows are the WMMA B columns)
// -------------------------------------------------------------------------
__global__ void prep_w_kernel(const float* __restrict__ w, _Float16* __restrict__ wh) {
    int t = blockIdx.x * blockDim.x + threadIdx.x;
    if (t < CO_ * CK_) wh[t] = (_Float16)w[t];
}

// -------------------------------------------------------------------------
// KNN: G = Xn * Xn^T fused with per-row top-9 (largest gram == nearest).
// One wave per 16-row tile; single-wave workgroup (barriers -> S_NOP).
// -------------------------------------------------------------------------
__global__ void __launch_bounds__(32) knn_kernel(const _Float16* __restrict__ xnh,
                                                 int* __restrict__ nn_idx,
                                                 int* __restrict__ ei) {
    __shared__ float gt[16][17];    // 16x16 gram tile (padded)
    __shared__ float mv[32][KNN_];  // per-lane top-9 values
    __shared__ int   mi[32][KNN_];  // per-lane top-9 indices

    const int lane = threadIdx.x & 31;
    const int wg   = blockIdx.x;                  // 0 .. B*(N/16)-1
    const int b    = wg / (N_ / 16);
    const int rowbase = (wg % (N_ / 16)) * 16;
    const _Float16* xb = xnh + (size_t)b * N_ * C_;

    const int m  = lane & 15;
    const int hi = lane >> 4;

    // A fragments: 16 rows x K=192 (6 blocks of 32). ISA 16-bit A layout:
    // lane m holds K chunks [koff..koff+7] and [koff+16..koff+23], koff=hi*8.
    v16h a[6];
    {
        const _Float16* rp = xb + (size_t)(rowbase + m) * C_;
        const int koff = hi * 8;
        #pragma unroll
        for (int kb = 0; kb < 6; ++kb) {
            v8h c0 = *(const v8h*)(rp + kb * 32 + koff);
            v8h c1 = *(const v8h*)(rp + kb * 32 + 16 + koff);
            #pragma unroll
            for (int q = 0; q < 8; ++q) { a[kb][q] = c0[q]; a[kb][8 + q] = c1[q]; }
        }
    }

    float kv[KNN_]; int ki[KNN_];
    #pragma unroll
    for (int i = 0; i < KNN_; ++i) { kv[i] = -1e30f; ki[i] = 0; }

    const int kb2 = hi * 8;    // B layout matches A: lane half hi holds K [hi*8..+7] and [16+hi*8..+7]
    for (int ct = 0; ct < N_ / 16; ++ct) {
        const int colbase = ct * 16;
        const _Float16* cp = xb + (size_t)(colbase + m) * C_;  // this lane's column point

        // Prefetch next tile's B data (global_prefetch_b8) while we compute.
        if (ct + 1 < N_ / 16)
            __builtin_prefetch(cp + (size_t)16 * C_, 0, 3);

        // Load ALL 6 B fragments (one clause, one wait), then the WMMA chain
        // runs back-to-back. The asm barrier keeps all fragments live so the
        // scheduler cannot re-serialize load->wait->wmma.
        v16h bf[6];
        #pragma unroll
        for (int kb = 0; kb < 6; ++kb) {
            v8h c0 = *(const v8h*)(cp + kb * 32 + kb2);
            v8h c1 = *(const v8h*)(cp + kb * 32 + kb2 + 16);
            #pragma unroll
            for (int q = 0; q < 8; ++q) { bf[kb][q] = c0[q]; bf[kb][8 + q] = c1[q]; }
        }
        FRAG_BARRIER6(bf);
        v8f c = {};
        #pragma unroll
        for (int kb = 0; kb < 6; ++kb)
            c = __builtin_amdgcn_wmma_f32_16x16x32_f16(false, a[kb], false, bf[kb],
                                                       (short)0, c, false, false);

        // C/D layout: lanes 0-15: M=v, N=lane; lanes 16-31: M=v+8, N=lane-16.
        #pragma unroll
        for (int v = 0; v < 8; ++v) gt[hi ? v + 8 : v][m] = c[v];
        __syncthreads();   // single-wave WG: S_NOP barrier, keeps LDS ordering
        // 2 lanes per row, 8 candidate columns each; sorted-descending insert.
        #pragma unroll
        for (int j = 0; j < 8; ++j) {
            const int n  = hi * 8 + j;
            const float val = gt[m][n];
            const int gi = colbase + n;
            if (val > kv[KNN_ - 1]) {
                kv[KNN_ - 1] = val; ki[KNN_ - 1] = gi;
                #pragma unroll
                for (int s = KNN_ - 1; s > 0; --s) {
                    if (kv[s] > kv[s - 1]) {
                        float tv = kv[s]; kv[s] = kv[s - 1]; kv[s - 1] = tv;
                        int   ti = ki[s]; ki[s] = ki[s - 1]; ki[s - 1] = ti;
                    }
                }
            }
        }
        __syncthreads();
    }

    #pragma unroll
    for (int i = 0; i < KNN_; ++i) { mv[lane][i] = kv[i]; mi[lane][i] = ki[i]; }
    __syncthreads();

    if (lane < 16) {
        const int r = lane;
        const int rowg = rowbase + r;
        int ia = 0, ib = 0;
        int* dst = nn_idx + ((size_t)b * N_ + rowg) * KNN_;
        int* e0  = ei + ((size_t)b * N_ + rowg) * KNN_;
        int* e1  = e0 + (size_t)B_ * N_ * KNN_;
        for (int i = 0; i < KNN_; ++i) {
            float va = mv[r][ia], vb = mv[r + 16][ib];
            int pick;
            if (va >= vb) { pick = mi[r][ia]; ++ia; }
            else          { pick = mi[r + 16][ib]; ++ib; }
            dst[i] = pick; e0[i] = pick; e1[i] = rowg;
        }
    }
}

// -------------------------------------------------------------------------
// Edge conv: one wave per point; 16 edge rows (9 real + 7 duplicates of row 0
// so max over the padded tile == max over K). z = feat*W^T + bias, BN, GELU,
// max over rows, scatter to (B, CO, N). B fragments are consumed in two
// batches of 6 (loads clustered via the asm barrier).
// -------------------------------------------------------------------------
__device__ __forceinline__ void load_feat8(const float* __restrict__ xi,
                                           const float* __restrict__ xj,
                                           int k0, float* t) {
    if (k0 < C_) {
        const float4* p = (const float4*)(xi + k0);
        float4 u0 = p[0], u1 = p[1];
        t[0]=u0.x; t[1]=u0.y; t[2]=u0.z; t[3]=u0.w;
        t[4]=u1.x; t[5]=u1.y; t[6]=u1.z; t[7]=u1.w;
    } else {
        const int kk = k0 - C_;
        const float4* pj = (const float4*)(xj + kk);
        const float4* pi = (const float4*)(xi + kk);
        float4 j0 = pj[0], j1 = pj[1], i0 = pi[0], i1 = pi[1];
        t[0]=j0.x-i0.x; t[1]=j0.y-i0.y; t[2]=j0.z-i0.z; t[3]=j0.w-i0.w;
        t[4]=j1.x-i1.x; t[5]=j1.y-i1.y; t[6]=j1.z-i1.z; t[7]=j1.w-i1.w;
    }
}

__global__ void __launch_bounds__(128) edge_conv_kernel(
        const float* __restrict__ xt, const _Float16* __restrict__ wh,
        const int* __restrict__ nn_idx,
        const float* __restrict__ bias, const float* __restrict__ gamma,
        const float* __restrict__ beta, float* __restrict__ out) {
    const int lane = threadIdx.x & 31;
    const int wid  = threadIdx.x >> 5;
    const int wg   = blockIdx.x * 4 + wid;   // point id, 0..B*N-1
    const int b = wg / N_, n = wg % N_;
    const int m  = lane & 15;
    const int hi = lane >> 4;

    const int mm = (m < KNN_) ? m : 0;                        // pad rows duplicate row 0
    const int j  = nn_idx[((size_t)b * N_ + n) * KNN_ + mm];
    const float* xi = xt + ((size_t)b * N_ + n) * C_;
    const float* xj = xt + ((size_t)b * N_ + j) * C_;

    // A fragments: 16 rows x K=384 (12 blocks), built from [x_i | x_j - x_i].
    const int koff = hi * 8;
    v16h a[12];
    #pragma unroll
    for (int kb = 0; kb < 12; ++kb) {
        float t1[8], t2[8];
        load_feat8(xi, xj, kb * 32 + koff,      t1);
        load_feat8(xi, xj, kb * 32 + koff + 16, t2);
        #pragma unroll
        for (int q = 0; q < 8; ++q) {
            a[kb][q]     = (_Float16)t1[q];
            a[kb][8 + q] = (_Float16)t2[q];
        }
    }

    const float bnscale = 0.9999950000375f;   // 1/sqrt(1+1e-5)
    for (int ot = 0; ot < CO_ / 16; ++ot) {
        const int oc = ot * 16 + m;           // this lane's output channel (N index)
        const float bi = bias[oc];
        const float sc = gamma[oc] * bnscale;
        const float bt = beta[oc];
        const _Float16* wrow = wh + (size_t)oc * CK_ + hi * 8;

        v8f c = {};
        #pragma unroll
        for (int g = 0; g < 2; ++g) {
            v16h bf[6];
            #pragma unroll
            for (int kb = 0; kb < 6; ++kb) {
                const _Float16* wp = wrow + (g * 6 + kb) * 32;
                v8h c0 = *(const v8h*)(wp);
                v8h c1 = *(const v8h*)(wp + 16);
                #pragma unroll
                for (int q = 0; q < 8; ++q) { bf[kb][q] = c0[q]; bf[kb][8 + q] = c1[q]; }
            }
            FRAG_BARRIER6(bf);
            #pragma unroll
            for (int kb = 0; kb < 6; ++kb)
                c = __builtin_amdgcn_wmma_f32_16x16x32_f16(false, a[g * 6 + kb], false,
                                                           bf[kb], (short)0, c,
                                                           false, false);
        }

        // bias + BN + exact GELU, then max over M (8 VGPRs local, + lane-half swap)
        float mx = -1e30f;
        #pragma unroll
        for (int v = 0; v < 8; ++v) {
            float z = c[v] + bi;
            z = z * sc + bt;
            float g = 0.5f * z * (1.f + erff(z * 0.70710678118654752f));
            mx = fmaxf(mx, g);
        }
        mx = fmaxf(mx, __shfl_xor(mx, 16, 32));
        if (lane < 16)
            out[((size_t)b * CO_ + (ot * 16 + lane)) * N_ + n] = mx;
    }
}

// -------------------------------------------------------------------------
extern "C" void kernel_launch(void* const* d_in, const int* in_sizes, int n_in,
                              void* d_out, int out_size, void* d_ws, size_t ws_size,
                              hipStream_t stream) {
    const float* x      = (const float*)d_in[0];
    const float* conv_w = (const float*)d_in[1];
    const float* conv_b = (const float*)d_in[2];
    const float* gamma  = (const float*)d_in[3];
    const float* beta   = (const float*)d_in[4];

    char* ws = (char*)d_ws;
    float*    xt  = (float*)(ws);                   //  9,633,792 B  (B,N,C) fp32
    _Float16* xnh = (_Float16*)(ws + 9633792);      //  4,816,896 B  normalized f16
    _Float16* wh  = (_Float16*)(ws + 14450688);     //    294,912 B  weights f16
    int*      nni = (int*)(ws + 14745600);          //    451,584 B  nn indices

    float* outf = (float*)d_out;                    // (B, CO, H, W)
    int*   ei   = (int*)(outf + (size_t)B_ * CO_ * N_);  // (2, B, N, K) int32

    hipLaunchKernelGGL(prep_points_kernel, dim3((B_ * N_ + 255) / 256), dim3(256),
                       0, stream, x, xt, xnh);
    hipLaunchKernelGGL(prep_w_kernel, dim3((CO_ * CK_ + 255) / 256), dim3(256),
                       0, stream, conv_w, wh);
    hipLaunchKernelGGL(knn_kernel, dim3(B_ * (N_ / 16)), dim3(32),
                       0, stream, xnh, nni, ei);
    hipLaunchKernelGGL(edge_conv_kernel, dim3(B_ * N_ / 4), dim3(128),
                       0, stream, xt, wh, nni, conv_b, gamma, beta, outf);
}
